// QKVAttention_7988639171103
// MI455X (gfx1250) — matrix operand order repaired and checked by reading, not compiled, not run on hardware
//
#include <hip/hip_runtime.h>

// Problem constants (B=8, N=8, T=512, C=256, H_DIM=512, heads=8, dh=64)
#define BN     64          // B*N
#define TSEQ   512
#define CIN    256
#define HDIM   512
#define NHEAD  8
#define DH     64
#define MROWS  (BN * TSEQ) // 32768 flattened tokens

typedef __bf16 bf16_t;
typedef __attribute__((ext_vector_type(16))) __bf16 v16bf;
typedef __attribute__((ext_vector_type(8)))  __bf16 v8bf;
typedef __attribute__((ext_vector_type(8)))  float  v8f;

union AFrag { v16bf v; v8bf h[2]; };

static __device__ __forceinline__ bf16_t f2bf(float f) {
  unsigned u = __builtin_bit_cast(unsigned, f);
  u += 0x7FFFu + ((u >> 16) & 1u);               // round-to-nearest-even
  unsigned short s = (unsigned short)(u >> 16);
  return __builtin_bit_cast(bf16_t, s);
}

static __device__ __forceinline__ v8f wmma_bf16(v16bf a, v16bf b, v8f c) {
  return __builtin_amdgcn_wmma_f32_16x16x32_bf16(
      /*neg_a=*/false, a, /*neg_b=*/false, b,
      /*c_mod=*/(short)0, c, /*reuse_a=*/false, /*reuse_b=*/false);
}

// A-fragment 16x32 (MxK), row-major source, rows m0..m0+15, K cols kk..kk+31.
// Layout: lane(r,hl): VGPR0-3 = K kk+hl*16..+7, VGPR4-7 = K kk+hl*16+8..+15.
static __device__ __forceinline__ v16bf load_a(const bf16_t* src, int ld,
                                               int m0, int kk, int r, int hl) {
  AFrag f;
  const bf16_t* p = src + (size_t)(m0 + r) * ld + kk + hl * 16;
  f.h[0] = *(const v8bf*)(p);
  f.h[1] = *(const v8bf*)(p + 8);
  return f.v;
}

// B-fragment 32x16 (KxN) where B[k][n] = src[n][k] (src row-major over n).
// Layout (same k order as load_a): lane(r,hl) = column n0+r, K run kk+hl*16..+15 (contiguous).
static __device__ __forceinline__ v16bf load_b(const bf16_t* src, int ld,
                                               int n0, int kk, int r, int hl) {
  return *(const v16bf*)(src + (size_t)(n0 + r) * ld + kk + hl * 16);
}

// ---------------------------------------------------------------- cvt f32->bf16
__global__ __launch_bounds__(256) void cvt_kernel(const float* __restrict__ src,
                                                  bf16_t* __restrict__ dst, int n) {
  int i = blockIdx.x * blockDim.x + threadIdx.x;
  if (i < n) dst[i] = f2bf(src[i]);
}

// ------------------------------------------------------- QKV projection (WMMA)
// Each wave: 32x64 output tile, register double-buffered K pipeline.
// z = 0/1/2 -> q/k/v.  q,k stored (token, HDIM) bf16.  v stored transposed:
// vt[(h*BN + bn)*DH + d][t]  (leading dim TSEQ) so PV B-fragments are contiguous.
__global__ __launch_bounds__(128) void qkv_kernel(
    const bf16_t* __restrict__ xb,
    const bf16_t* __restrict__ Wqb, const bf16_t* __restrict__ Wkb,
    const bf16_t* __restrict__ Wvb,
    bf16_t* __restrict__ qb, bf16_t* __restrict__ kb, bf16_t* __restrict__ vtb) {
  const int wave = threadIdx.x >> 5, lane = threadIdx.x & 31;
  const int r = lane & 15, hl = lane >> 4;
  const int m0 = (blockIdx.x * 4 + wave) * 32;   // 1024 row tiles of 32
  const int n0 = blockIdx.y * 64;                // 8 column groups of 64
  const int z  = blockIdx.z;
  const bf16_t* W = (z == 0) ? Wqb : (z == 1) ? Wkb : Wvb;

  v8f acc[2][4] = {};
  v16bf a0 = load_a(xb, CIN, m0,      0, r, hl);
  v16bf a1 = load_a(xb, CIN, m0 + 16, 0, r, hl);
  v16bf b0 = load_b(W, CIN, n0,      0, r, hl);
  v16bf b1 = load_b(W, CIN, n0 + 16, 0, r, hl);
  v16bf b2 = load_b(W, CIN, n0 + 32, 0, r, hl);
  v16bf b3 = load_b(W, CIN, n0 + 48, 0, r, hl);

#pragma unroll
  for (int kk = 0; kk < CIN; kk += 32) {
    v16bf na0, na1, nb0, nb1, nb2, nb3;
    if (kk + 32 < CIN) {                         // issue next-K loads first
      na0 = load_a(xb, CIN, m0,      kk + 32, r, hl);
      na1 = load_a(xb, CIN, m0 + 16, kk + 32, r, hl);
      nb0 = load_b(W, CIN, n0,      kk + 32, r, hl);
      nb1 = load_b(W, CIN, n0 + 16, kk + 32, r, hl);
      nb2 = load_b(W, CIN, n0 + 32, kk + 32, r, hl);
      nb3 = load_b(W, CIN, n0 + 48, kk + 32, r, hl);
    }
    acc[0][0] = wmma_bf16(a0, b0, acc[0][0]);
    acc[1][0] = wmma_bf16(a1, b0, acc[1][0]);
    acc[0][1] = wmma_bf16(a0, b1, acc[0][1]);
    acc[1][1] = wmma_bf16(a1, b1, acc[1][1]);
    acc[0][2] = wmma_bf16(a0, b2, acc[0][2]);
    acc[1][2] = wmma_bf16(a1, b2, acc[1][2]);
    acc[0][3] = wmma_bf16(a0, b3, acc[0][3]);
    acc[1][3] = wmma_bf16(a1, b3, acc[1][3]);
    if (kk + 32 < CIN) {
      a0 = na0; a1 = na1; b0 = nb0; b1 = nb1; b2 = nb2; b3 = nb3;
    }
  }

  if (z < 2) {                                   // q, k: (token, HDIM)
    bf16_t* dst = (z == 0) ? qb : kb;
#pragma unroll
    for (int sub = 0; sub < 2; ++sub)
#pragma unroll
      for (int i = 0; i < 4; ++i) {
        int col = n0 + i * 16 + r;
#pragma unroll
        for (int j = 0; j < 8; ++j)
          dst[(size_t)(m0 + sub * 16 + hl * 8 + j) * HDIM + col] = f2bf(acc[sub][i][j]);
      }
  } else {                                       // v: transposed per head
    int bn = m0 / TSEQ, t0 = m0 % TSEQ;          // 32-row tile lies within one (b,n)
#pragma unroll
    for (int sub = 0; sub < 2; ++sub)
#pragma unroll
      for (int i = 0; i < 4; ++i) {
        int colH = n0 + i * 16 + r;
        int h = colH >> 6, d = colH & 63;
        v8bf pk;
#pragma unroll
        for (int j = 0; j < 8; ++j) pk[j] = f2bf(acc[sub][i][j]);
        *(v8bf*)(vtb + (((size_t)(h * BN + bn) * DH + d) * TSEQ
                        + t0 + sub * 16 + hl * 8)) = pk;
      }
  }
}

// --------------------------------------------- flash attention (one wave/tile)
__global__ __launch_bounds__(128) void attn_kernel(
    const bf16_t* __restrict__ qb, const bf16_t* __restrict__ kb,
    const bf16_t* __restrict__ vtb, bf16_t* __restrict__ hob) {
  __shared__ bf16_t probs_s[4][16 * 32];         // 1 KB per wave
  const int wave = threadIdx.x >> 5, lane = threadIdx.x & 31;
  const int r = lane & 15, hl = lane >> 4;
  bf16_t* probs = probs_s[wave];

  const int idx = blockIdx.x * 4 + wave;         // 16384 wave-tiles
  const int tt  = idx & 31;                      // q-tile within T
  const int hbn = idx >> 5;
  const int h   = hbn >> 6, bn = hbn & 63;
  const int t0  = tt * 16;

  const bf16_t* qrow = qb + (size_t)bn * TSEQ * HDIM + (size_t)h * DH;
  const bf16_t* krow = kb + (size_t)bn * TSEQ * HDIM + (size_t)h * DH;
  const bf16_t* vt   = vtb + (size_t)(h * BN + bn) * DH * TSEQ;

  // q fragments (16 rows x dh=64) held in registers for the whole s-loop
  v16bf q0 = load_a(qrow, HDIM, t0, 0,  r, hl);
  v16bf q1 = load_a(qrow, HDIM, t0, 32, r, hl);

  float mrow[8], lrow[8];
#pragma unroll
  for (int j = 0; j < 8; ++j) { mrow[j] = -3.0e38f; lrow[j] = 0.f; }
  v8f o[4] = {};
  const float scale = 0.125f;                    // 1/sqrt(dh)

  for (int s0 = 0; s0 < TSEQ; s0 += 32) {
    // pull next s-step's k rows / v rows toward the WGP (global_prefetch)
    if (s0 + 32 < TSEQ) {
      __builtin_prefetch(krow + (size_t)(s0 + 32 + r) * HDIM, 0, 3);
      __builtin_prefetch(krow + (size_t)(s0 + 48 + r) * HDIM, 0, 3);
#pragma unroll
      for (int i = 0; i < 4; ++i)
        __builtin_prefetch(vt + (size_t)(i * 16 + r) * TSEQ + s0 + 32, 0, 3);
    }

    v8f e0 = {}, e1 = {};
    v16bf kb00 = load_b(krow, HDIM, s0,      0,  r, hl);
    v16bf kb01 = load_b(krow, HDIM, s0,      32, r, hl);
    v16bf kb10 = load_b(krow, HDIM, s0 + 16, 0,  r, hl);
    v16bf kb11 = load_b(krow, HDIM, s0 + 16, 32, r, hl);
    e0 = wmma_bf16(q0, kb00, e0);
    e0 = wmma_bf16(q1, kb01, e0);
    e1 = wmma_bf16(q0, kb10, e1);
    e1 = wmma_bf16(q1, kb11, e1);

#pragma unroll
    for (int j = 0; j < 8; ++j) {
      float a0 = e0[j] * scale, a1 = e1[j] * scale;
      // row max over the 32 columns: slot j is a fixed row across each lane-half
      float v = fmaxf(a0, a1);
      v = fmaxf(v, __shfl_xor(v, 1, 32));
      v = fmaxf(v, __shfl_xor(v, 2, 32));
      v = fmaxf(v, __shfl_xor(v, 4, 32));
      v = fmaxf(v, __shfl_xor(v, 8, 32));
      float mnew = fmaxf(mrow[j], v);
      float corr = __expf(mrow[j] - mnew);
      float p0 = __expf(a0 - mnew);
      float p1 = __expf(a1 - mnew);
      probs[(hl * 8 + j) * 32 + r]      = f2bf(p0);   // D-layout -> LDS
      probs[(hl * 8 + j) * 32 + 16 + r] = f2bf(p1);
      float ps = p0 + p1;
      ps += __shfl_xor(ps, 1, 32);
      ps += __shfl_xor(ps, 2, 32);
      ps += __shfl_xor(ps, 4, 32);
      ps += __shfl_xor(ps, 8, 32);
      lrow[j] = lrow[j] * corr + ps;
      mrow[j] = mnew;
      o[0][j] *= corr; o[1][j] *= corr; o[2][j] *= corr; o[3][j] *= corr;
    }
    __builtin_amdgcn_wave_barrier();               // order LDS stores vs loads
    v16bf pa = load_a(probs, 32, 0, 0, r, hl);     // LDS -> A-layout (16x32)
    __builtin_amdgcn_wave_barrier();
#pragma unroll
    for (int i = 0; i < 4; ++i) {
      v16bf vb = load_b(vt, TSEQ, i * 16, s0, r, hl);   // vt[d][s], contiguous s
      o[i] = wmma_bf16(pa, vb, o[i]);
    }
  }

  // normalize and emit head_out (token, HDIM) bf16, head h at columns h*64..
#pragma unroll
  for (int i = 0; i < 4; ++i) {
    int colH = h * DH + i * 16 + r;
#pragma unroll
    for (int j = 0; j < 8; ++j) {
      float inv = 1.0f / lrow[j];
      int t = t0 + hl * 8 + j;
      hob[((size_t)bn * TSEQ + t) * HDIM + colH] = f2bf(o[i][j] * inv);
    }
  }
}

// ----------------------------------------------- output projection (+bias, f32)
// Each wave: 32x64 output tile, register double-buffered K pipeline (K=512).
__global__ __launch_bounds__(128) void proj_kernel(
    const bf16_t* __restrict__ hob, const bf16_t* __restrict__ Wpb,
    const float* __restrict__ bp, float* __restrict__ out) {
  const int wave = threadIdx.x >> 5, lane = threadIdx.x & 31;
  const int r = lane & 15, hl = lane >> 4;
  const int m0 = (blockIdx.x * 4 + wave) * 32;
  const int n0 = blockIdx.y * 64;

  v8f acc[2][4] = {};
  v16bf a0 = load_a(hob, HDIM, m0,      0, r, hl);
  v16bf a1 = load_a(hob, HDIM, m0 + 16, 0, r, hl);
  v16bf b0 = load_b(Wpb, HDIM, n0,      0, r, hl);
  v16bf b1 = load_b(Wpb, HDIM, n0 + 16, 0, r, hl);
  v16bf b2 = load_b(Wpb, HDIM, n0 + 32, 0, r, hl);
  v16bf b3 = load_b(Wpb, HDIM, n0 + 48, 0, r, hl);

#pragma unroll 4
  for (int kk = 0; kk < HDIM; kk += 32) {
    v16bf na0, na1, nb0, nb1, nb2, nb3;
    if (kk + 32 < HDIM) {
      na0 = load_a(hob, HDIM, m0,      kk + 32, r, hl);
      na1 = load_a(hob, HDIM, m0 + 16, kk + 32, r, hl);
      nb0 = load_b(Wpb, HDIM, n0,      kk + 32, r, hl);
      nb1 = load_b(Wpb, HDIM, n0 + 16, kk + 32, r, hl);
      nb2 = load_b(Wpb, HDIM, n0 + 32, kk + 32, r, hl);
      nb3 = load_b(Wpb, HDIM, n0 + 48, kk + 32, r, hl);
    }
    acc[0][0] = wmma_bf16(a0, b0, acc[0][0]);
    acc[1][0] = wmma_bf16(a1, b0, acc[1][0]);
    acc[0][1] = wmma_bf16(a0, b1, acc[0][1]);
    acc[1][1] = wmma_bf16(a1, b1, acc[1][1]);
    acc[0][2] = wmma_bf16(a0, b2, acc[0][2]);
    acc[1][2] = wmma_bf16(a1, b2, acc[1][2]);
    acc[0][3] = wmma_bf16(a0, b3, acc[0][3]);
    acc[1][3] = wmma_bf16(a1, b3, acc[1][3]);
    if (kk + 32 < HDIM) {
      a0 = na0; a1 = na1; b0 = nb0; b1 = nb1; b2 = nb2; b3 = nb3;
    }
  }

#pragma unroll
  for (int sub = 0; sub < 2; ++sub)
#pragma unroll
    for (int i = 0; i < 4; ++i) {
      int col = n0 + i * 16 + r;
      float bias = bp[col];
#pragma unroll
      for (int j = 0; j < 8; ++j)
        out[(size_t)(m0 + sub * 16 + hl * 8 + j) * HDIM + col] = acc[sub][i][j] + bias;
    }
}

// ----------------------------------------------------------------------- launch
extern "C" void kernel_launch(void* const* d_in, const int* in_sizes, int n_in,
                              void* d_out, int out_size, void* d_ws, size_t ws_size,
                              hipStream_t stream) {
  (void)in_sizes; (void)n_in; (void)out_size; (void)ws_size;
  const float* x  = (const float*)d_in[0];
  const float* Wq = (const float*)d_in[1];
  const float* Wk = (const float*)d_in[2];
  const float* Wv = (const float*)d_in[3];
  const float* Wp = (const float*)d_in[4];
  const float* bp = (const float*)d_in[5];
  float* out = (float*)d_out;

  bf16_t* ws = (bf16_t*)d_ws;
  const size_t NX  = (size_t)MROWS * CIN;          // 8,388,608
  const size_t NW  = (size_t)HDIM * CIN;           //   131,072
  const size_t NWP = (size_t)HDIM * HDIM;          //   262,144
  const size_t NQ  = (size_t)MROWS * HDIM;         // 16,777,216
  bf16_t* xb  = ws;
  bf16_t* Wqb = xb  + NX;
  bf16_t* Wkb = Wqb + NW;
  bf16_t* Wvb = Wkb + NW;
  bf16_t* Wpb = Wvb + NW;
  bf16_t* qb  = Wpb + NWP;
  bf16_t* kb  = qb  + NQ;
  bf16_t* vtb = kb  + NQ;
  bf16_t* hob = vtb + NQ;

  cvt_kernel<<<dim3((unsigned)((NX  + 255) / 256)), 256, 0, stream>>>(x,  xb,  (int)NX);
  cvt_kernel<<<dim3((unsigned)((NW  + 255) / 256)), 256, 0, stream>>>(Wq, Wqb, (int)NW);
  cvt_kernel<<<dim3((unsigned)((NW  + 255) / 256)), 256, 0, stream>>>(Wk, Wkb, (int)NW);
  cvt_kernel<<<dim3((unsigned)((NW  + 255) / 256)), 256, 0, stream>>>(Wv, Wvb, (int)NW);
  cvt_kernel<<<dim3((unsigned)((NWP + 255) / 256)), 256, 0, stream>>>(Wp, Wpb, (int)NWP);

  // 1024 row-tiles of 32 / 4 waves, 8 column groups, 3 weights
  qkv_kernel<<<dim3(256, 8, 3), 128, 0, stream>>>(xb, Wqb, Wkb, Wvb, qb, kb, vtb);
  // 8 heads * 64 (b,n) * 32 q-tiles = 16384 wave-tiles / 4 waves
  attn_kernel<<<dim3(4096), 128, 0, stream>>>(qb, kb, vtb, hob);
  // 1024 row-tiles of 32 / 4 waves, 8 column groups
  proj_kernel<<<dim3(256, 8), 128, 0, stream>>>(hob, Wpb, bp, out);
}
